// CONV_tiny_add_partial_558345748883
// MI455X (gfx1250) — matrix operand order repaired and checked by reading, not compiled, not run on hardware
//
#include <hip/hip_runtime.h>
#include <hip/hip_bf16.h>

typedef __attribute__((ext_vector_type(16))) _Float16 v16h;
typedef __attribute__((ext_vector_type(8)))  float    v8f;

#define EPS 1e-5f

// ---------------------------------------------------------------------------
// WMMA f16 16x16x32 A-fragment element->K mapping used here (contiguous; ISA table order differs):
// lane L: m = L&15, half = L>>4. v16h element e (VGPR e/2, slot e%2):
//   K = e + 16*half (contiguous order, matching B)
// B-fragment (32x16): lane L: n = L&15, half = L>>4; v16h element e: K = e + 16*half
//   -> contiguous 32B read when K (=ci) is the fastest-varying dimension.
// C/D: VGPR d: lanes 0-15 hold M=d, lanes 16-31 hold M=d+8; N = lane&15.
// ---------------------------------------------------------------------------
__device__ __forceinline__ int a_k_of_e(int e, int half) {
  return e + 16 * half;
}

// Async copy: global -> LDS, 16 bytes per lane, tracked with ASYNCcnt.
__device__ __forceinline__ void async_copy_b128(uint32_t lds_off, const void* gptr) {
  uint64_t ga = (uint64_t)(uintptr_t)gptr;
  asm volatile("global_load_async_to_lds_b128 %0, %1, off"
               :: "v"(lds_off), "v"(ga) : "memory");
}
__device__ __forceinline__ void wait_async0() {
  asm volatile("s_wait_asynccnt 0x0" ::: "memory");
}
__device__ __forceinline__ uint32_t lds_offset_of(const void* p) {
  return (uint32_t)(uintptr_t)p;   // low 32 bits of flat shared address = LDS offset
}

// Pre-swizzle conv weights (COUT,32,5,5) fp32 -> fragment-ready f16:
// apack[mt][tap][lane][e]  (K-dim per tap == ci, CIN==32 exactly)
__global__ void pack_conv_w(const float* __restrict__ w, _Float16* __restrict__ apack,
                            int cout_total) {
  int idx = blockIdx.x * blockDim.x + threadIdx.x;
  int total = (cout_total / 16) * 25 * 32 * 16;
  if (idx >= total) return;
  int e    = idx & 15;
  int lane = (idx >> 4) & 31;
  int tap  = (idx >> 9) % 25;
  int mt   = (idx >> 9) / 25;
  int m = lane & 15, half = lane >> 4;
  int cout = mt * 16 + m;
  int ci   = a_k_of_e(e, half);
  apack[idx] = (_Float16)w[(cout * 32 + ci) * 25 + tap];
}

// Pre-swizzle fc weights (10,64,4,4) -> apack[ks][lane][e].
// K-dim flat order is (y,x,ci) to match NHWC x3: kg = (y*4+x)*64 + ci.
__global__ void pack_fc_w(const float* __restrict__ w4, _Float16* __restrict__ apack) {
  int idx = blockIdx.x * blockDim.x + threadIdx.x;
  if (idx >= 32 * 32 * 16) return;
  int e    = idx & 15;
  int lane = (idx >> 4) & 31;
  int ks   = idx >> 9;
  int m = lane & 15, half = lane >> 4;
  int kg  = ks * 32 + a_k_of_e(e, half);
  int ci  = kg & 63;        // channel
  int pos = kg >> 6;        // y*4+x
  _Float16 val = (_Float16)0.f;
  if (m < 10) val = (_Float16)w4[m * 1024 + ci * 16 + pos];
  apack[idx] = val;
}

// ---------------------------------------------------------------------------
// Layer 1: direct fp32 conv 3->32, 5x5 pad 2, fused BN + 2x2 avgpool + clip.
// Emits f16 NHWC (1024,16,16,32) so layer 2 can stage rows contiguously.
// ---------------------------------------------------------------------------
__global__ __launch_bounds__(128)
void conv1_direct(const float* __restrict__ x, const float* __restrict__ w1,
                  const float* __restrict__ g, const float* __restrict__ bb,
                  const float* __restrict__ mm, const float* __restrict__ vv,
                  _Float16* __restrict__ xout) {
  const int H = 32, W = 32, CIN = 3, COUT = 32;
  __shared__ float xs[6 * 36 * 3];       // [rr][col(+halo)][ci]
  __shared__ float wsh[32 * 75];
  __shared__ float sinv[32], sbeta[32];
  int tid = threadIdx.x;
  int b  = blockIdx.x >> 4;
  int yp = blockIdx.x & 15;
  int y0 = yp * 2;
  for (int i = tid; i < 6 * 36 * 3; i += 128) {
    int rr  = i / (36 * 3);
    int rem = i % (36 * 3);
    int c = rem / 3, ci = rem % 3;
    int row = y0 - 2 + rr, xx = c - 2;
    float val = 0.f;
    if (row >= 0 && row < H && xx >= 0 && xx < W)
      val = x[((b * CIN + ci) * H + row) * W + xx];
    xs[i] = val;
  }
  for (int i = tid; i < 32 * 75; i += 128) wsh[i] = w1[i];
  if (tid < 32) {
    float inv = g[tid] * rsqrtf(vv[tid] + EPS);
    sinv[tid] = inv;
    sbeta[tid] = bb[tid] - mm[tid] * inv;
  }
  __syncthreads();
  for (int o = tid; o < 32 * 16; o += 128) {
    int cout = o & 31;         // cout fastest across lanes -> coalesced NHWC store
    int xp   = o >> 5;
    float acc = 0.f;
    for (int r = 0; r < 2; ++r)
      for (int c2 = 0; c2 < 2; ++c2) {
        float conv = 0.f;
        for (int ci = 0; ci < 3; ++ci)
          #pragma unroll
          for (int ky = 0; ky < 5; ++ky)
            #pragma unroll
            for (int kx = 0; kx < 5; ++kx)
              conv += wsh[(cout * 3 + ci) * 25 + ky * 5 + kx] *
                      xs[((r + ky) * 36 + (2 * xp + c2 + kx)) * 3 + ci];
        acc += conv * sinv[cout] + sbeta[cout];   // BN affine: BN then pool == pool then BN
      }
    acc *= 0.25f;
    acc = fminf(fmaxf(acc, 0.f), 1.f);
    xout[((b * 16 + yp) * 16 + xp) * COUT + cout] = (_Float16)acc;   // NHWC
  }
}

// ---------------------------------------------------------------------------
// Layers 2/3: implicit-GEMM 5x5 pad-2 conv via v_wmma_f32_16x16x32_f16.
// Input/output are NHWC f16. One workgroup = (image, pooled row): 2 conv rows
// x all channels; 4 waves, each one 16x16 tile, 25 WMMAs (K = CIN = 32).
// Staging: zero-fill LDS (halo/OOB) then async b128 copies global->LDS
// (contiguous NHWC rows), s_wait_asynccnt before barrier.
// ---------------------------------------------------------------------------
template<int CIN, int COUT, int H, int W>
__global__ __launch_bounds__(128)
void conv5x5_wmma(const _Float16* __restrict__ xin, const _Float16* __restrict__ apack,
                  const float* __restrict__ g, const float* __restrict__ bb,
                  const float* __restrict__ mm, const float* __restrict__ vv,
                  _Float16* __restrict__ xout) {
  static_assert(CIN == 32, "K per tap must be 32");
  constexpr int NT = (2 * W) / 16;           // N tiles over 2 rows x W cols
  constexpr int SW = W + 4;                  // staged width with halo
  __shared__ __align__(32) _Float16 xs[6 * SW * CIN];  // [rr][col][ci]
  __shared__ float cbuf[COUT * 2 * W];
  __shared__ float sinv[COUT], sbeta[COUT];
  int tid = threadIdx.x;
  int b  = blockIdx.x / (H / 2);
  int yp = blockIdx.x % (H / 2);
  int y0 = yp * 2;

  // 1) zero-fill the whole staging buffer (covers halo cols + OOB rows)
  {
    float4 z = {0.f, 0.f, 0.f, 0.f};
    float4* p = (float4*)xs;
    constexpr int NZ = 6 * SW * CIN * 2 / 16;
    for (int i = tid; i < NZ; i += 128) p[i] = z;
  }
  for (int i = tid; i < COUT; i += 128) {
    float inv = g[i] * rsqrtf(vv[i] + EPS);
    sinv[i] = inv;
    sbeta[i] = bb[i] - mm[i] * inv;
  }
  __syncthreads();

  // 2) async-copy valid input rows (NHWC: one row = W*CIN contiguous f16)
  {
    constexpr int CPR = W * CIN / 8;         // b128 chunks per row
    for (int i = tid; i < 6 * CPR; i += 128) {
      int rr = i / CPR, c = i % CPR;
      int row = y0 - 2 + rr;
      if (row >= 0 && row < H) {
        uint32_t lds = lds_offset_of(&xs[(rr * SW + 2) * CIN + c * 8]);
        const _Float16* gp = xin + ((size_t)(b * H + row) * W) * CIN + c * 8;
        async_copy_b128(lds, gp);
      }
    }
  }
  wait_async0();
  __syncthreads();

  // 3) WMMA main loop
  int wave = tid >> 5, lane = tid & 31;
  int mtile = wave / NT, ntile = wave % NT;
  int half = lane >> 4;
  int p = ntile * 16 + (lane & 15);          // pixel index within the 2xW patch
  int r = p / W, xx = p % W;
  v8f acc = {0.f, 0.f, 0.f, 0.f, 0.f, 0.f, 0.f, 0.f};
  const v16h* ap = (const v16h*)apack + (mtile * 25) * 32 + lane;
  #pragma unroll
  for (int ky = 0; ky < 5; ++ky) {
    #pragma unroll
    for (int kx = 0; kx < 5; ++kx) {
      v16h a = ap[(ky * 5 + kx) * 32];                                          // global b128 x2
      v16h bf = *(const v16h*)(&xs[((r + ky) * SW + (xx + kx)) * CIN + half * 16]); // ds b128 x2
      acc = __builtin_amdgcn_wmma_f32_16x16x32_f16(false, a, false, bf,
                                                   (short)0, acc, false, false);
    }
  }
  // 4) BN epilogue into LDS tile
  #pragma unroll
  for (int d = 0; d < 8; ++d) {
    int cout = mtile * 16 + d + 8 * half;
    cbuf[(cout * 2 + r) * W + xx] = acc[d] * sinv[cout] + sbeta[cout];
  }
  __syncthreads();
  // 5) 2x2 avg-pool + clip -> f16 NHWC (cout fastest across lanes -> coalesced)
  for (int o = tid; o < COUT * (W / 2); o += 128) {
    int cout = o % COUT, xp = o / COUT;
    float s = cbuf[(cout * 2 + 0) * W + 2 * xp] + cbuf[(cout * 2 + 0) * W + 2 * xp + 1]
            + cbuf[(cout * 2 + 1) * W + 2 * xp] + cbuf[(cout * 2 + 1) * W + 2 * xp + 1];
    s *= 0.25f;
    s = fminf(fmaxf(s, 0.f), 1.f);
    xout[((b * (H / 2) + yp) * (W / 2) + xp) * COUT + cout] = (_Float16)s;
  }
}

// ---------------------------------------------------------------------------
// Layer 4: fc as GEMM (M=10 pad 16, N=16 batches/wave, K=1024 = 32 k-steps).
// x3 is NHWC so per-batch features are flat in (y,x,ci) order -> each lane's
// 16 f16 are one contiguous 32B global load. Fused BN1d, fp32 out.
// ---------------------------------------------------------------------------
__global__ __launch_bounds__(32)
void fc_wmma(const _Float16* __restrict__ x3, const _Float16* __restrict__ apack,
             const float* __restrict__ g, const float* __restrict__ bb,
             const float* __restrict__ mm, const float* __restrict__ vv,
             float* __restrict__ out) {
  int lane = threadIdx.x & 31;
  int half = lane >> 4;
  int n    = lane & 15;
  int bidx = blockIdx.x * 16 + n;
  v8f acc = {0.f, 0.f, 0.f, 0.f, 0.f, 0.f, 0.f, 0.f};
  const v16h* ap = (const v16h*)apack + lane;
  const v16h* bp = (const v16h*)(x3 + (size_t)bidx * 1024);
  #pragma unroll
  for (int ks = 0; ks < 32; ++ks) {
    v16h a  = ap[ks * 32];
    v16h bf = bp[ks * 2 + half];
    acc = __builtin_amdgcn_wmma_f32_16x16x32_f16(false, a, false, bf,
                                                 (short)0, acc, false, false);
  }
  #pragma unroll
  for (int d = 0; d < 8; ++d) {
    int m = d + 8 * half;
    if (m < 10) {
      float inv = g[m] * rsqrtf(vv[m] + EPS);
      out[bidx * 10 + m] = acc[d] * inv + (bb[m] - mm[m] * inv);
    }
  }
}

// ---------------------------------------------------------------------------
extern "C" void kernel_launch(void* const* d_in, const int* in_sizes, int n_in,
                              void* d_out, int out_size, void* d_ws, size_t ws_size,
                              hipStream_t stream) {
  const float* x_in = (const float*)d_in[0];
  const float* w1   = (const float*)d_in[1];
  const float* w2   = (const float*)d_in[2];
  const float* w3   = (const float*)d_in[3];
  const float* w4   = (const float*)d_in[4];
  const float* g1 = (const float*)d_in[5],  *b1 = (const float*)d_in[6],
             * m1 = (const float*)d_in[7],  *v1 = (const float*)d_in[8];
  const float* g2 = (const float*)d_in[9],  *b2 = (const float*)d_in[10],
             * m2 = (const float*)d_in[11], *v2 = (const float*)d_in[12];
  const float* g3 = (const float*)d_in[13], *b3 = (const float*)d_in[14],
             * m3 = (const float*)d_in[15], *v3 = (const float*)d_in[16];
  const float* g4 = (const float*)d_in[17], *b4 = (const float*)d_in[18],
             * m4 = (const float*)d_in[19], *v4 = (const float*)d_in[20];

  // workspace carve-up (~23.3 MB total, all 256B-aligned)
  char* ws = (char*)d_ws;
  size_t off = 0;
  auto carve = [&](size_t bytes) -> void* {
    void* p = ws + off;
    off = (off + bytes + 255) & ~(size_t)255;
    return p;
  };
  _Float16* x1  = (_Float16*)carve(1024ull * 16 * 16 * 32 * 2);  // NHWC
  _Float16* x2  = (_Float16*)carve(1024ull * 8  * 8  * 32 * 2);  // NHWC
  _Float16* x3  = (_Float16*)carve(1024ull * 4  * 4  * 64 * 2);  // NHWC
  _Float16* ap2 = (_Float16*)carve(2ull  * 25 * 32 * 16 * 2);
  _Float16* ap3 = (_Float16*)carve(4ull  * 25 * 32 * 16 * 2);
  _Float16* ap4 = (_Float16*)carve(32ull * 32 * 16 * 2);

  pack_conv_w<<<(2 * 25 * 32 * 16 + 255) / 256, 256, 0, stream>>>(w2, ap2, 32);
  pack_conv_w<<<(4 * 25 * 32 * 16 + 255) / 256, 256, 0, stream>>>(w3, ap3, 64);
  pack_fc_w  <<<(32 * 32 * 16 + 255) / 256, 256, 0, stream>>>(w4, ap4);

  conv1_direct<<<1024 * 16, 128, 0, stream>>>(x_in, w1, g1, b1, m1, v1, x1);
  conv5x5_wmma<32, 32, 16, 16><<<1024 * 8, 128, 0, stream>>>(x1, ap2, g2, b2, m2, v2, x2);
  conv5x5_wmma<32, 64,  8,  8><<<1024 * 4, 128, 0, stream>>>(x2, ap3, g3, b3, m3, v3, x3);
  fc_wmma<<<64, 32, 0, stream>>>(x3, ap4, g4, b4, m4, v4, (float*)d_out);
}
